// xgnn_poly_force_full_23313082483599
// MI455X (gfx1250) — matrix operand order repaired and checked by reading, not compiled, not run on hardware
//
#include <hip/hip_runtime.h>

typedef __attribute__((ext_vector_type(16))) __bf16 v16bf;
typedef __attribute__((ext_vector_type(8)))  __bf16 v8bf;
typedef __attribute__((ext_vector_type(8)))  float  v8f;

// ---------------- problem constants ----------------
#define Gn    32
#define An    24
#define DEGn  8
#define Nn    768
#define En    6144
#define Tn    43008
#define MATn  169
#define EMBn  128
#define RBFDn 16
#define SBFDn 7
#define SBFn  112
#define HEADSn 16
#define CHn   256
#define HDn   16
#define Ln    4
#define EPGn  192          /* edges per graph = An*DEGn */

#define MATp  192          /* MATn padded to 32 */
#define RBFp  32           /* RBFDn padded to 32 */
#define SBFp  128          /* SBFn padded to 32 */

#define CUTf  5.0f
#define PIf   3.14159265358979f
#define SQ2C  0.632455532034f   /* sqrt(2/CUT) */
#define INVS  0.25f             /* 1/sqrt(HD) */

// ---------------- device helpers ----------------
__device__ __forceinline__ float sigm (float z){ return 1.f/(1.f+__expf(-z)); }
__device__ __forceinline__ float siluf(float z){ return z*sigm(z); }
__device__ __forceinline__ float dsilu(float z){ float s=sigm(z); return s*(1.f+z*(1.f-s)); }
__device__ __forceinline__ float envel(float d){
  float u=d/CUTf; if(u>=1.f) return 0.f;
  float u2=u*u,u4=u2*u2,u5=u4*u,u6=u5*u,u7=u6*u;
  return 1.f-21.f*u5+35.f*u6-15.f*u7;
}
__device__ __forceinline__ float denvel(float d){
  float u=d/CUTf; if(u>=1.f) return 0.f;
  float u2=u*u,u4=u2*u2,u5=u4*u,u6=u5*u;
  return (-105.f*u4+210.f*u5-105.f*u6)/CUTf;
}

// =====================================================================
// Operand packing to zero-padded bf16
//   A: row-major [Mp x Kp];  B: N-major [Ncp x Kp], Ncp padded to 64
// =====================================================================
__global__ void k_pack_a(const float* __restrict__ A, int lda, int M, int K,
                         __bf16* __restrict__ out, int Kp, int total)
{
  int i = blockIdx.x*blockDim.x + threadIdx.x;
  if (i >= total) return;
  int m = i / Kp, k = i % Kp;
  float v = (m < M && k < K) ? A[(size_t)m*lda + k] : 0.f;
  out[i] = (__bf16)v;
}

__global__ void k_pack_b(const float* __restrict__ B, int ldb, int tb, int K, int Nc,
                         __bf16* __restrict__ out, int Kp, int total)
{
  int i = blockIdx.x*blockDim.x + threadIdx.x;
  if (i >= total) return;
  int n = i / Kp, k = i % Kp;
  float v = 0.f;
  if (k < K && n < Nc) v = tb ? B[(size_t)n*ldb + k] : B[(size_t)k*ldb + n];
  out[i] = (__bf16)v;
}

// =====================================================================
// bf16-WMMA matmul on packed operands: one wave per 16x64 tile
// (1 M-tile x 4 N-tiles -> A fragment reused by 4 WMMAs per k-step).
// Inner loop: 2x16B A loads + 4x32B B loads + 4x v_wmma, branch-free.
// mode 0: C=z   1: C=silu(z)   2: C=resid+silu(z)   3: C+=z
// =====================================================================
__global__ void k_mm(const __bf16* __restrict__ Abf, const __bf16* __restrict__ Bbf,
                     const float* __restrict__ bias, const float* __restrict__ resid,
                     float* __restrict__ C, int ldc, float* __restrict__ Z, int ldz,
                     int M, int Kp, int Nc, int mode)
{
  const int lane = threadIdx.x;
  const int half = lane >> 4;
  const int l15  = lane & 15;
  const int m0 = blockIdx.x * 16;
  const int n0 = blockIdx.y * 64;
  const int row = m0 + l15;
  const __bf16* arow = Abf + (size_t)row*Kp;
  const __bf16* bp[4];
#pragma unroll
  for (int j = 0; j < 4; ++j) bp[j] = Bbf + (size_t)(n0 + 16*j + l15)*Kp;

  v8f acc[4] = {};
#pragma unroll 2
  for (int kk = 0; kk < Kp; kk += 32) {
    // A 16x32: lane = M row, slots = contiguous K (hi lanes +16), matching B
    v8bf alo = *(const v8bf*)(arow + kk + 16*half);
    v8bf ahi = *(const v8bf*)(arow + kk + 16*half + 8);
    v16bf a = __builtin_shufflevector(alo, ahi, 0,1,2,3,4,5,6,7,8,9,10,11,12,13,14,15);
    __builtin_prefetch(arow + kk + 64, 0, 1);          // global_prefetch_b8
#pragma unroll
    for (int j = 0; j < 4; ++j) {
      // B 32x16: lane = N column, slots = contiguous K (hi lanes +16)
      v16bf b = *(const v16bf*)(bp[j] + kk + 16*half);
      acc[j] = __builtin_amdgcn_wmma_f32_16x16x32_bf16(false, a, false, b,
                                                       (short)0, acc[j], false, false);
    }
  }

#pragma unroll
  for (int j = 0; j < 4; ++j) {
    int col = n0 + 16*j + l15;
    float bn = (bias && col < Nc) ? bias[col] : 0.f;
#pragma unroll
    for (int r = 0; r < 8; ++r) {
      int m = m0 + 8*half + r;          // C/D: VGPR r -> M=r (+8 for hi lanes)
      if (m >= M || col >= Nc) continue;
      float z = acc[j][r] + bn;
      size_t ci = (size_t)m*ldc + col;
      if (Z) Z[(size_t)m*ldz + col] = z;
      float o;
      if      (mode == 0) o = z;
      else if (mode == 1) o = siluf(z);
      else if (mode == 2) o = resid[ci] + siluf(z);
      else                o = C[ci] + z;
      C[ci] = o;
    }
  }
}

// ---------------- elementwise / geometry kernels ----------------
__global__ void k_clear(float* p, int n){
  int i = blockIdx.x*blockDim.x + threadIdx.x;
  if (i < n) p[i] = 0.f;
}

__global__ void k_edge_geom_fwd(const int* __restrict__ eidx,
                                const float* __restrict__ pos,
                                float* __restrict__ dbuf, float* __restrict__ envbuf,
                                __bf16* __restrict__ rbfenv /* [E x RBFp] bf16 */)
{
  int e = blockIdx.x*blockDim.x + threadIdx.x;
  if (e >= En) return;
  int s = eidx[e], t = eidx[En + e];
  float dx = pos[s*3+0]-pos[t*3+0];
  float dy = pos[s*3+1]-pos[t*3+1];
  float dz = pos[s*3+2]-pos[t*3+2];
  float d = sqrtf(dx*dx+dy*dy+dz*dz);
  float ev = envel(d);
  dbuf[e] = d; envbuf[e] = ev;
  float invd = 1.f/d;
#pragma unroll
  for (int r = 0; r < RBFDn; ++r) {
    float nn = (float)(r+1);
    float rb = SQ2C * __sinf(nn*PIf*d/CUTf) * invd;
    rbfenv[e*RBFp + r] = (__bf16)(rb * ev);
  }
#pragma unroll
  for (int r = RBFDn; r < RBFp; ++r) rbfenv[e*RBFp + r] = (__bf16)0.f;
}

__global__ void k_scale_env(const float* __restrict__ ea, const float* __restrict__ env,
                            __bf16* __restrict__ out /* [E x MATp] bf16 */)
{
  int i = blockIdx.x*blockDim.x + threadIdx.x;
  if (i >= En*MATp) return;
  int e = i / MATp, m = i % MATp;
  float v = (m < MATn) ? ea[(size_t)e*MATn + m] * env[e] : 0.f;
  out[i] = (__bf16)v;
}

__global__ void k_node_emb(const int* __restrict__ x, const float* __restrict__ tbl,
                           __bf16* __restrict__ out /* [N x EMBn] bf16 */)
{
  int i = blockIdx.x*blockDim.x + threadIdx.x;
  if (i >= Nn*EMBn) return;
  int n = i / EMBn, c = i % EMBn;
  out[i] = (__bf16)tbl[x[n]*EMBn + c];
}

__global__ void k_triplet_fwd(const int* __restrict__ ai, const int* __restrict__ aj,
                              const int* __restrict__ ak, const float* __restrict__ pos,
                              const int* __restrict__ srcE,
                              const float* __restrict__ dbuf, const float* __restrict__ envbuf,
                              float* __restrict__ angbuf,
                              __bf16* __restrict__ sbf /* [T x SBFp] bf16 */)
{
  int t = blockIdx.x*blockDim.x + threadIdx.x;
  if (t >= Tn) return;
  int i = ai[t], j = aj[t], k = ak[t];
  float jix = pos[i*3+0]-pos[j*3+0], jiy = pos[i*3+1]-pos[j*3+1], jiz = pos[i*3+2]-pos[j*3+2];
  float jkx = pos[k*3+0]-pos[j*3+0], jky = pos[k*3+1]-pos[j*3+1], jkz = pos[k*3+2]-pos[j*3+2];
  float wx = jiy*jkz - jiz*jky, wy = jiz*jkx - jix*jkz, wz = jix*jky - jiy*jkx;
  float s = sqrtf(wx*wx+wy*wy+wz*wz);
  float c = jix*jkx + jiy*jky + jiz*jkz;
  float ang = atan2f(s, c);
  angbuf[t] = ang;
  int se = srcE[t];
  float ds = dbuf[se], ev = envbuf[se], invd = 1.f/ds;
  float rad[RBFDn];
#pragma unroll
  for (int r = 0; r < RBFDn; ++r) {
    float nn = (float)(r+1);
    rad[r] = SQ2C * __sinf(nn*PIf*ds/CUTf) * invd * ev;
  }
#pragma unroll
  for (int l = 0; l < SBFDn; ++l) {
    float cl = __cosf((float)l * ang);
#pragma unroll
    for (int r = 0; r < RBFDn; ++r)
      sbf[(size_t)t*SBFp + l*RBFDn + r] = (__bf16)(cl * rad[r]);
  }
#pragma unroll
  for (int p = SBFn; p < SBFp; ++p) sbf[(size_t)t*SBFp + p] = (__bf16)0.f;
}

// ---------------- attention (fixed 7-way contiguous segments) --------
__global__ void k_attn_fwd(const float* __restrict__ qb, const float* __restrict__ kb,
                           const float* __restrict__ vb, const float* __restrict__ esb,
                           const float* __restrict__ neat, const int* __restrict__ srcE,
                           float* __restrict__ agg, float* __restrict__ attn)
{
  int e = blockIdx.x, c = threadIdx.x;
  int head = c >> 4, l16 = c & 15;
  __shared__ float slog[7*HEADSn];
  float qc   = qb[(size_t)e*CHn + c];
  float eatc = neat[(size_t)(e >> 3)*CHn + c];   // atom_j = e/DEG
  float ve[7];
#pragma unroll
  for (int t = 0; t < 7; ++t) {
    size_t tt = (size_t)e*7 + t;
    int s = srcE[tt];
    float es = esb[tt*CHn + c];
    float kv = kb[(size_t)s*CHn + c]*es + eatc;
    ve[t] = vb[(size_t)s*CHn + c]*es;
    float p = qc * kv;
#pragma unroll
    for (int off = 8; off >= 1; off >>= 1) p += __shfl_xor(p, off, 16);
    if (l16 == 0) slog[t*HEADSn + head] = p * INVS;
  }
  __syncthreads();
  float mx = -1e30f;
#pragma unroll
  for (int t = 0; t < 7; ++t) mx = fmaxf(mx, slog[t*HEADSn + head]);
  float ex[7], den = 0.f;
#pragma unroll
  for (int t = 0; t < 7; ++t) { ex[t] = __expf(slog[t*HEADSn+head] - mx); den += ex[t]; }
  float inv = 1.f/(den + 1e-16f);
  float ac = 0.f;
#pragma unroll
  for (int t = 0; t < 7; ++t) {
    float w = ex[t]*inv;
    ac += w * ve[t];
    if (l16 == 0) attn[((size_t)e*7 + t)*HEADSn + head] = w;
  }
  agg[(size_t)e*CHn + c] = ac;
}

__global__ void k_attn_bwd(const float* __restrict__ qb, const float* __restrict__ kb,
                           const float* __restrict__ vb, float* __restrict__ esb, /* in: esb, out: g_esb */
                           const float* __restrict__ neat, const int* __restrict__ srcE,
                           const float* __restrict__ attn, const float* __restrict__ gagg,
                           float* __restrict__ gq, float* __restrict__ gk, float* __restrict__ gv)
{
  int e = blockIdx.x, c = threadIdx.x;
  int head = c >> 4;
  float qc   = qb[(size_t)e*CHn + c];
  float eatc = neat[(size_t)(e >> 3)*CHn + c];
  float gc   = gagg[(size_t)e*CHn + c];
  float es[7], ks[7], vs[7], at[7], ga[7]; int ss[7];
#pragma unroll
  for (int t = 0; t < 7; ++t) {
    size_t tt = (size_t)e*7 + t;
    int s = srcE[tt]; ss[t] = s;
    es[t] = esb[tt*CHn + c];
    ks[t] = kb[(size_t)s*CHn + c];
    vs[t] = vb[(size_t)s*CHn + c];
    at[t] = attn[tt*HEADSn + head];
    float p = gc * vs[t] * es[t];
#pragma unroll
    for (int off = 8; off >= 1; off >>= 1) p += __shfl_xor(p, off, 16);
    ga[t] = p;                                  // dL/d attn_t (all lanes)
  }
  float S = 0.f;
#pragma unroll
  for (int t = 0; t < 7; ++t) S += at[t]*ga[t];
  float gqc = 0.f;
#pragma unroll
  for (int t = 0; t < 7; ++t) {
    float gl = at[t]*(ga[t] - S);               // softmax backward
    gqc += INVS*gl*(ks[t]*es[t] + eatc);
    atomicAdd(&gk[(size_t)ss[t]*CHn + c], INVS*gl*qc*es[t]);
    atomicAdd(&gv[(size_t)ss[t]*CHn + c], at[t]*gc*es[t]);
    esb[((size_t)e*7 + t)*CHn + c] = at[t]*gc*vs[t] + INVS*gl*qc*ks[t];  // g_esb
  }
  gq[(size_t)e*CHn + c] = gqc;
}

// ---------------- output head / energy ----------------
__global__ void k_energy(const float* __restrict__ t2, const float* __restrict__ W2,
                         const float* __restrict__ b2, float* __restrict__ out)
{
  int g = blockIdx.x, tid = threadIdx.x;
  __shared__ float red[256];
  float acc = 0.f;
  for (int idx = tid; idx < EPGn*EMBn; idx += 256) {
    int e = g*EPGn + idx/EMBn, c = idx % EMBn;
    acc += t2[(size_t)e*EMBn + c] * W2[c];
  }
  red[tid] = acc; __syncthreads();
  for (int o = 128; o >= 1; o >>= 1) { if (tid < o) red[tid] += red[tid+o]; __syncthreads(); }
  if (tid == 0) out[g] = red[0] + (float)EPGn * b2[0];
}

__global__ void k_gz1(const float* __restrict__ z1, const float* __restrict__ W2,
                      float* __restrict__ gz)
{
  int i = blockIdx.x*blockDim.x + threadIdx.x;
  if (i >= En*EMBn) return;
  gz[i] = W2[i % EMBn] * dsilu(z1[i]);
}

__global__ void k_silu_bwd(const float* __restrict__ gy, const float* __restrict__ z,
                           float* __restrict__ gz, int n)
{
  int i = blockIdx.x*blockDim.x + threadIdx.x;
  if (i >= n) return;
  gz[i] = gy[i] * dsilu(z[i]);
}

__global__ void k_split_bwd(const float* __restrict__ gcat, const float* __restrict__ zmat,
                            const float* __restrict__ zrbf,
                            float* __restrict__ gzmat, float* __restrict__ gzrbf)
{
  int i = blockIdx.x*blockDim.x + threadIdx.x;
  if (i >= En*EMBn) return;
  int e = i / EMBn, c = i % EMBn;
  gzmat[i] = gcat[(size_t)e*CHn + c]        * dsilu(zmat[i]);
  gzrbf[i] = gcat[(size_t)e*CHn + EMBn + c] * dsilu(zrbf[i]);
}

// ---------------- geometry backward ----------------
__global__ void k_triplet_bwd(const int* __restrict__ ai, const int* __restrict__ aj,
                              const int* __restrict__ ak, const float* __restrict__ pos,
                              const int* __restrict__ srcE,
                              const float* __restrict__ dbuf, const float* __restrict__ angbuf,
                              const float* __restrict__ gsbf,
                              float* __restrict__ gd, float* __restrict__ gpos)
{
  int t = blockIdx.x*blockDim.x + threadIdx.x;
  if (t >= Tn) return;
  float ang = angbuf[t];
  int se = srcE[t];
  float ds = dbuf[se], ev = envel(ds), evp = denvel(ds), invd = 1.f/ds;
  float rad[RBFDn], drad[RBFDn];
#pragma unroll
  for (int r = 0; r < RBFDn; ++r) {
    float nn = (float)(r+1);
    float arg = nn*PIf*ds/CUTf;
    float s_ = __sinf(arg), c_ = __cosf(arg);
    float rb  = SQ2C * s_ * invd;
    float drb = SQ2C * ((nn*PIf/CUTf)*c_*invd - s_*invd*invd);
    rad[r]  = rb*ev;
    drad[r] = drb*ev + rb*evp;
  }
  float gr[RBFDn]; float gang = 0.f;
#pragma unroll
  for (int r = 0; r < RBFDn; ++r) gr[r] = 0.f;
#pragma unroll
  for (int l = 0; l < SBFDn; ++l) {
    float cl = __cosf((float)l*ang), sl = __sinf((float)l*ang);
#pragma unroll
    for (int r = 0; r < RBFDn; ++r) {
      float gs = gsbf[(size_t)t*SBFn + l*RBFDn + r];
      gr[r] += gs*cl;
      gang  += gs*rad[r]*(-(float)l*sl);
    }
  }
  float g_d = 0.f;
#pragma unroll
  for (int r = 0; r < RBFDn; ++r) g_d += gr[r]*drad[r];
  atomicAdd(&gd[se], g_d);

  // angle gradient -> positions
  int i = ai[t], j = aj[t], k = ak[t];
  float jix = pos[i*3+0]-pos[j*3+0], jiy = pos[i*3+1]-pos[j*3+1], jiz = pos[i*3+2]-pos[j*3+2];
  float jkx = pos[k*3+0]-pos[j*3+0], jky = pos[k*3+1]-pos[j*3+1], jkz = pos[k*3+2]-pos[j*3+2];
  float wx = jiy*jkz - jiz*jky, wy = jiz*jkx - jix*jkz, wz = jix*jky - jiy*jkx;
  float s2 = wx*wx+wy*wy+wz*wz;
  float s = sqrtf(s2);
  float c = jix*jkx + jiy*jky + jiz*jkz;
  float n2 = s2 + c*c;
  if (s > 1e-9f && n2 > 1e-20f) {
    float cs = gang * c/(n2*s);   // coeff for d s
    float cc = -gang * s/n2;      // coeff for d c
    // ds/dji = (jk x w)/s ; ds/djk = (w x ji)/s
    float ax = jky*wz - jkz*wy, ay = jkz*wx - jkx*wz, az = jkx*wy - jky*wx;
    float bx = wy*jiz - wz*jiy, by = wz*jix - wx*jiz, bz = wx*jiy - wy*jix;
    float gix = cs*ax + cc*jkx, giy = cs*ay + cc*jky, giz = cs*az + cc*jkz;
    float gkx2 = cs*bx + cc*jix, gky2 = cs*by + cc*jiy, gkz2 = cs*bz + cc*jiz;
    atomicAdd(&gpos[i*3+0], gix); atomicAdd(&gpos[i*3+1], giy); atomicAdd(&gpos[i*3+2], giz);
    atomicAdd(&gpos[k*3+0], gkx2); atomicAdd(&gpos[k*3+1], gky2); atomicAdd(&gpos[k*3+2], gkz2);
    atomicAdd(&gpos[j*3+0], -(gix+gkx2)); atomicAdd(&gpos[j*3+1], -(giy+gky2));
    atomicAdd(&gpos[j*3+2], -(giz+gkz2));
  }
}

__global__ void k_edge_geom_bwd(const int* __restrict__ eidx, const float* __restrict__ pos,
                                const float* __restrict__ ea,
                                const float* __restrict__ dbuf, const float* __restrict__ envbuf,
                                const float* __restrict__ geaenv, const float* __restrict__ grbfenv,
                                const float* __restrict__ gd_sbf,
                                float* __restrict__ geattr, float* __restrict__ gpos)
{
  int e = blockIdx.x*blockDim.x + threadIdx.x;
  if (e >= En) return;
  float d = dbuf[e], ev = envbuf[e], evp = denvel(d), invd = 1.f/d;
  float genv = 0.f;
  for (int m = 0; m < MATn; ++m) {
    float ge = geaenv[(size_t)e*MATn + m];
    genv += ge * ea[(size_t)e*MATn + m];
    geattr[(size_t)e*MATn + m] = ge * ev;     // dL/d edge_attr
  }
  float g_d = 0.f;
#pragma unroll
  for (int r = 0; r < RBFDn; ++r) {
    float nn = (float)(r+1);
    float arg = nn*PIf*d/CUTf;
    float s_ = __sinf(arg), c_ = __cosf(arg);
    float rb  = SQ2C * s_ * invd;
    float drb = SQ2C * ((nn*PIf/CUTf)*c_*invd - s_*invd*invd);
    float gr = grbfenv[e*RBFDn + r];
    genv += gr * rb;
    g_d  += gr * ev * drb;
  }
  g_d += genv * evp;
  g_d += gd_sbf[e];
  int s = eidx[e], t = eidx[En + e];
  float ux = (pos[s*3+0]-pos[t*3+0])*invd;
  float uy = (pos[s*3+1]-pos[t*3+1])*invd;
  float uz = (pos[s*3+2]-pos[t*3+2])*invd;
  atomicAdd(&gpos[s*3+0],  g_d*ux); atomicAdd(&gpos[s*3+1],  g_d*uy); atomicAdd(&gpos[s*3+2],  g_d*uz);
  atomicAdd(&gpos[t*3+0], -g_d*ux); atomicAdd(&gpos[t*3+1], -g_d*uy); atomicAdd(&gpos[t*3+2], -g_d*uz);
}

// edge_attr_grad contraction: sgrads[g*A+a, c] = sum_{e in g} sum_m eag[e,a,c,m]*geattr[e,m]
__global__ void k_sgrads(const float* __restrict__ eag, const float* __restrict__ geattr,
                         float* __restrict__ sg)
{
  int b = blockIdx.x, tid = threadIdx.x;
  int g = b/(An*3), rem = b%(An*3), a = rem/3, cd = rem%3;
  __shared__ float red[256];
  float acc = 0.f;
  for (int el = 0; el < EPGn; ++el) {
    int e = g*EPGn + el;
    size_t base = (((size_t)e*An + a)*3 + cd)*MATn;
    for (int m = tid; m < MATn; m += 256)
      acc += eag[base + m] * geattr[(size_t)e*MATn + m];
  }
  red[tid] = acc; __syncthreads();
  for (int o = 128; o >= 1; o >>= 1) { if (tid < o) red[tid] += red[tid+o]; __syncthreads(); }
  if (tid == 0) sg[(size_t)(g*An + a)*3 + cd] = red[0];
}

__global__ void k_force(const float* __restrict__ gpos, const float* __restrict__ sg,
                        float* __restrict__ out)
{
  int i = blockIdx.x*blockDim.x + threadIdx.x;
  if (i >= Nn*3) return;
  out[Gn + i] = -(gpos[i] + sg[i]);
}

// ---------------- host-side helpers ----------------
static inline int roundup(int x, int a){ return (x + a - 1)/a*a; }

static inline void pack_a(hipStream_t st, const float* A, int lda, int M, int K,
                          __bf16* out, int Kp)
{
  int Mp = roundup(M, 16);
  int total = Mp * Kp;
  k_pack_a<<<(total+255)/256, 256, 0, st>>>(A, lda, M, K, out, Kp, total);
}

static inline void pack_b(hipStream_t st, const float* B, int ldb, int tb, int K, int Nc,
                          __bf16* out, int Kp)
{
  int Ncp = roundup(Nc, 64);          // pad to 64 so 16x64 tiles stay in-bounds
  int total = Ncp * Kp;
  k_pack_b<<<(total+255)/256, 256, 0, st>>>(B, ldb, tb, K, Nc, out, Kp, total);
}

static inline void matmul(hipStream_t st, const __bf16* Abf, const __bf16* Bbf,
                          const float* bias, const float* resid,
                          float* C, int ldc, float* Z, int ldz,
                          int M, int Kp, int Nc, int mode)
{
  dim3 gr((M+15)/16, (Nc+63)/64);
  k_mm<<<gr, 32, 0, st>>>(Abf, Bbf, bias, resid, C, ldc, Z, ldz, M, Kp, Nc, mode);
}

extern "C" void kernel_launch(void* const* d_in, const int* in_sizes, int n_in,
                              void* d_out, int out_size, void* d_ws, size_t ws_size,
                              hipStream_t stream)
{
  (void)in_sizes; (void)n_in; (void)out_size; (void)ws_size;
  const int*   x_sp = (const int*)  d_in[0];
  const int*   eidx = (const int*)  d_in[1];   // [2,E]
  const int*   nidx = (const int*)  d_in[2];   // [2,T], row0 = srcE
  const int*   ai   = (const int*)  d_in[3];
  const int*   aj   = (const int*)  d_in[4];
  const int*   ak   = (const int*)  d_in[5];
  const float* pos  = (const float*)d_in[8];
  const float* ea   = (const float*)d_in[9];
  const float* eag  = (const float*)d_in[10];
  const float* embT = (const float*)d_in[11];
  const float* matW = (const float*)d_in[12];
  const float* matB = (const float*)d_in[13];
  const float* rbfW = (const float*)d_in[14];
  const float* rbfB = (const float*)d_in[15];
  const float* emW  = (const float*)d_in[16];
  const float* emB  = (const float*)d_in[17];
  const float* inW  = (const float*)d_in[18];
  const float* inB  = (const float*)d_in[19];
  const float* Wq   = (const float*)d_in[20];
  const float* Wk   = (const float*)d_in[21];
  const float* Wv   = (const float*)d_in[22];
  const float* Wo   = (const float*)d_in[23];
  const float* bo   = (const float*)d_in[24];
  const float* Wsb  = (const float*)d_in[25];
  const float* Wea  = (const float*)d_in[26];
  const float* o1W  = (const float*)d_in[27];
  const float* o1B  = (const float*)d_in[28];
  const float* o2W  = (const float*)d_in[29];
  const float* o2B  = (const float*)d_in[30];
  float* out = (float*)d_out;
  const int* srcE = nidx;   // row 0

  // ------- workspace layout (float units, 64B-aligned chunks) -------
  float* W0 = (float*)d_ws;
  size_t off = 0;
  auto alloc = [&](size_t nfloats){
    float* p = W0 + off; off += roundup((int)nfloats, 16); return p; };
  auto allocbf = [&](size_t nhalves){
    float* p = W0 + off; off += roundup((int)((nhalves+1)/2), 16); return (__bf16*)p; };

  float*  dB    = alloc(En);
  float*  envB  = alloc(En);
  float*  cat   = alloc((size_t)En*CHn);
  float*  zmat  = alloc((size_t)En*EMBn);
  float*  zrbf  = alloc((size_t)En*EMBn);
  float*  zemb  = alloc((size_t)En*EMBn);
  float*  tbuf  = alloc((size_t)En*EMBn);
  float*  hB    = alloc((size_t)(Ln+1)*En*CHn);
  float*  zoB   = alloc((size_t)Ln*En*CHn);
  float*  angB  = alloc(Tn);
  float*  esb   = alloc((size_t)Tn*CHn);
  float*  neat  = alloc((size_t)Nn*CHn);
  float*  qB    = alloc((size_t)En*CHn);
  float*  kB    = alloc((size_t)En*CHn);
  float*  vB    = alloc((size_t)En*CHn);
  float*  agg   = alloc((size_t)En*CHn);
  float*  attnB = alloc((size_t)Ln*Tn*HEADSn);
  float*  z1    = alloc((size_t)En*EMBn);
  float*  t2    = alloc((size_t)En*EMBn);
  float*  gz1   = alloc((size_t)En*EMBn);
  float*  gh    = alloc((size_t)En*CHn);
  float*  gtmp  = alloc((size_t)En*CHn);
  float*  gagg  = alloc((size_t)En*CHn);
  float*  gq    = alloc((size_t)En*CHn);
  float*  gk    = alloc((size_t)En*CHn);
  float*  gv    = alloc((size_t)En*CHn);
  float*  gsbf  = alloc((size_t)Tn*SBFn);
  float*  gt    = alloc((size_t)En*EMBn);
  float*  gz2   = alloc((size_t)En*EMBn);
  float*  gcat  = alloc((size_t)En*CHn);
  float*  gzm   = alloc((size_t)En*EMBn);
  float*  gzr   = alloc((size_t)En*EMBn);
  float*  geae  = alloc((size_t)En*MATn);
  float*  geat  = alloc((size_t)En*MATn);
  float*  grbe  = alloc((size_t)En*RBFDn);
  float*  gd    = alloc(En);
  float*  gpos  = alloc((size_t)Nn*3);
  float*  sg    = alloc((size_t)Nn*3);
  // packed bf16 operands
  __bf16* eaenvP = allocbf((size_t)En*MATp);
  __bf16* rbfeP  = allocbf((size_t)En*RBFp);
  __bf16* sbfP   = allocbf((size_t)Tn*SBFp);
  __bf16* nembP  = allocbf((size_t)Nn*EMBn);
  __bf16* Apack  = allocbf((size_t)Tn*CHn);     // largest A operand (T x 256)
  __bf16* Bpack  = allocbf((size_t)CHn*CHn);    // largest weight (256 x 256)

  // ================= forward =================
  k_edge_geom_fwd<<<(En+127)/128, 128, 0, stream>>>(eidx, pos, dB, envB, rbfeP);
  k_scale_env<<<(En*MATp+255)/256, 256, 0, stream>>>(ea, envB, eaenvP);
  k_node_emb<<<(Nn*EMBn+255)/256, 256, 0, stream>>>(x_sp, embT, nembP);
  k_triplet_fwd<<<(Tn+127)/128, 128, 0, stream>>>(ai, aj, ak, pos, srcE, dB, envB, angB, sbfP);

  // neo_x / rbf-MLP into concatenated [E,256] (cols 0:128 and 128:256)
  pack_b(stream, matW, EMBn, 0, MATn, EMBn, Bpack, MATp);
  matmul(stream, eaenvP, Bpack, matB, nullptr, cat,      CHn, zmat, EMBn, En, MATp, EMBn, 1);
  pack_b(stream, rbfW, EMBn, 0, RBFDn, EMBn, Bpack, RBFp);
  matmul(stream, rbfeP,  Bpack, rbfB, nullptr, cat+EMBn, CHn, zrbf, EMBn, En, RBFp, EMBn, 1);
  // embedding MLP + input projection
  pack_a(stream, cat, CHn, En, CHn, Apack, CHn);
  pack_b(stream, emW, EMBn, 0, CHn, EMBn, Bpack, CHn);
  matmul(stream, Apack, Bpack, emB, nullptr, tbuf, EMBn, zemb, EMBn, En, CHn, EMBn, 1);
  pack_a(stream, tbuf, EMBn, En, EMBn, Apack, EMBn);
  pack_b(stream, inW, CHn, 0, EMBn, CHn, Bpack, EMBn);
  matmul(stream, Apack, Bpack, inB, nullptr, hB, CHn, nullptr, 0, En, EMBn, CHn, 0);

  for (int l = 0; l < Ln; ++l) {
    float* hl = hB + (size_t)l*En*CHn;
    float* hn = hB + (size_t)(l+1)*En*CHn;
    float* zo = zoB + (size_t)l*En*CHn;
    pack_a(stream, hl, CHn, En, CHn, Apack, CHn);
    pack_b(stream, Wq + (size_t)l*CHn*CHn, CHn, 0, CHn, CHn, Bpack, CHn);
    matmul(stream, Apack, Bpack, nullptr, nullptr, qB, CHn, nullptr, 0, En, CHn, CHn, 0);
    pack_b(stream, Wk + (size_t)l*CHn*CHn, CHn, 0, CHn, CHn, Bpack, CHn);
    matmul(stream, Apack, Bpack, nullptr, nullptr, kB, CHn, nullptr, 0, En, CHn, CHn, 0);
    pack_b(stream, Wv + (size_t)l*CHn*CHn, CHn, 0, CHn, CHn, Bpack, CHn);
    matmul(stream, Apack, Bpack, nullptr, nullptr, vB, CHn, nullptr, 0, En, CHn, CHn, 0);
    pack_b(stream, Wsb + (size_t)l*SBFn*CHn, CHn, 0, SBFn, CHn, Bpack, SBFp);
    matmul(stream, sbfP, Bpack, nullptr, nullptr, esb, CHn, nullptr, 0, Tn, SBFp, CHn, 0);
    pack_b(stream, Wea + (size_t)l*EMBn*CHn, CHn, 0, EMBn, CHn, Bpack, EMBn);
    matmul(stream, nembP, Bpack, nullptr, nullptr, neat, CHn, nullptr, 0, Nn, EMBn, CHn, 0);
    k_attn_fwd<<<En, CHn, 0, stream>>>(qB, kB, vB, esb, neat, srcE, agg,
                                       attnB + (size_t)l*Tn*HEADSn);
    pack_a(stream, agg, CHn, En, CHn, Apack, CHn);
    pack_b(stream, Wo + (size_t)l*CHn*CHn, CHn, 0, CHn, CHn, Bpack, CHn);
    matmul(stream, Apack, Bpack, bo + (size_t)l*CHn, hl, hn, CHn, zo, CHn, En, CHn, CHn, 2);
  }
  float* h4 = hB + (size_t)Ln*En*CHn;
  pack_a(stream, h4, CHn, En, CHn, Apack, CHn);
  pack_b(stream, o1W, EMBn, 0, CHn, EMBn, Bpack, CHn);
  matmul(stream, Apack, Bpack, o1B, nullptr, t2, EMBn, z1, EMBn, En, CHn, EMBn, 1);
  k_energy<<<Gn, 256, 0, stream>>>(t2, o2W, o2B, out);

  // ================= backward =================
  k_gz1<<<(En*EMBn+255)/256, 256, 0, stream>>>(z1, o2W, gz1);
  pack_a(stream, gz1, EMBn, En, EMBn, Apack, EMBn);
  pack_b(stream, o1W, EMBn, 1, EMBn, CHn, Bpack, EMBn);          // out1_W^T
  matmul(stream, Apack, Bpack, nullptr, nullptr, gh, CHn, nullptr, 0, En, EMBn, CHn, 0);

  k_clear<<<(Tn*SBFn+255)/256, 256, 0, stream>>>(gsbf, Tn*SBFn);
  for (int l = Ln-1; l >= 0; --l) {
    float* hl = hB + (size_t)l*En*CHn;
    float* zo = zoB + (size_t)l*En*CHn;
    k_silu_bwd<<<(En*CHn+255)/256, 256, 0, stream>>>(gh, zo, gtmp, En*CHn);
    pack_a(stream, gtmp, CHn, En, CHn, Apack, CHn);
    pack_b(stream, Wo + (size_t)l*CHn*CHn, CHn, 1, CHn, CHn, Bpack, CHn);  // Wo^T
    matmul(stream, Apack, Bpack, nullptr, nullptr, gagg, CHn, nullptr, 0, En, CHn, CHn, 0);
    // recompute q,k,v,esb,neat for this layer
    pack_a(stream, hl, CHn, En, CHn, Apack, CHn);
    pack_b(stream, Wq + (size_t)l*CHn*CHn, CHn, 0, CHn, CHn, Bpack, CHn);
    matmul(stream, Apack, Bpack, nullptr, nullptr, qB, CHn, nullptr, 0, En, CHn, CHn, 0);
    pack_b(stream, Wk + (size_t)l*CHn*CHn, CHn, 0, CHn, CHn, Bpack, CHn);
    matmul(stream, Apack, Bpack, nullptr, nullptr, kB, CHn, nullptr, 0, En, CHn, CHn, 0);
    pack_b(stream, Wv + (size_t)l*CHn*CHn, CHn, 0, CHn, CHn, Bpack, CHn);
    matmul(stream, Apack, Bpack, nullptr, nullptr, vB, CHn, nullptr, 0, En, CHn, CHn, 0);
    pack_b(stream, Wsb + (size_t)l*SBFn*CHn, CHn, 0, SBFn, CHn, Bpack, SBFp);
    matmul(stream, sbfP, Bpack, nullptr, nullptr, esb, CHn, nullptr, 0, Tn, SBFp, CHn, 0);
    pack_b(stream, Wea + (size_t)l*EMBn*CHn, CHn, 0, EMBn, CHn, Bpack, EMBn);
    matmul(stream, nembP, Bpack, nullptr, nullptr, neat, CHn, nullptr, 0, Nn, EMBn, CHn, 0);
    k_clear<<<(En*CHn+255)/256, 256, 0, stream>>>(gk, En*CHn);
    k_clear<<<(En*CHn+255)/256, 256, 0, stream>>>(gv, En*CHn);
    k_attn_bwd<<<En, CHn, 0, stream>>>(qB, kB, vB, esb, neat, srcE,
                                       attnB + (size_t)l*Tn*HEADSn, gagg, gq, gk, gv);
    // g_h_l = g_h_{l+1} + gq@Wq^T + gk@Wk^T + gv@Wv^T   (accumulate)
    pack_a(stream, gq, CHn, En, CHn, Apack, CHn);
    pack_b(stream, Wq + (size_t)l*CHn*CHn, CHn, 1, CHn, CHn, Bpack, CHn);
    matmul(stream, Apack, Bpack, nullptr, nullptr, gh, CHn, nullptr, 0, En, CHn, CHn, 3);
    pack_a(stream, gk, CHn, En, CHn, Apack, CHn);
    pack_b(stream, Wk + (size_t)l*CHn*CHn, CHn, 1, CHn, CHn, Bpack, CHn);
    matmul(stream, Apack, Bpack, nullptr, nullptr, gh, CHn, nullptr, 0, En, CHn, CHn, 3);
    pack_a(stream, gv, CHn, En, CHn, Apack, CHn);
    pack_b(stream, Wv + (size_t)l*CHn*CHn, CHn, 1, CHn, CHn, Bpack, CHn);
    matmul(stream, Apack, Bpack, nullptr, nullptr, gh, CHn, nullptr, 0, En, CHn, CHn, 3);
    // g_edge_sbf += g_esb @ Wsbf^T  (esb now holds g_esb)
    pack_a(stream, esb, CHn, Tn, CHn, Apack, CHn);
    pack_b(stream, Wsb + (size_t)l*SBFn*CHn, CHn, 1, CHn, SBFn, Bpack, CHn);
    matmul(stream, Apack, Bpack, nullptr, nullptr, gsbf, SBFn, nullptr, 0, Tn, CHn, SBFn, 3);
  }

  // input MLP backward
  pack_a(stream, gh, CHn, En, CHn, Apack, CHn);
  pack_b(stream, inW, CHn, 1, CHn, EMBn, Bpack, CHn);            // in_W^T
  matmul(stream, Apack, Bpack, nullptr, nullptr, gt, EMBn, nullptr, 0, En, CHn, EMBn, 0);
  k_silu_bwd<<<(En*EMBn+255)/256, 256, 0, stream>>>(gt, zemb, gz2, En*EMBn);
  pack_a(stream, gz2, EMBn, En, EMBn, Apack, EMBn);
  pack_b(stream, emW, EMBn, 1, EMBn, CHn, Bpack, EMBn);          // embt_W^T
  matmul(stream, Apack, Bpack, nullptr, nullptr, gcat, CHn, nullptr, 0, En, EMBn, CHn, 0);
  k_split_bwd<<<(En*EMBn+255)/256, 256, 0, stream>>>(gcat, zmat, zrbf, gzm, gzr);
  pack_a(stream, gzm, EMBn, En, EMBn, Apack, EMBn);
  pack_b(stream, matW, EMBn, 1, EMBn, MATn, Bpack, EMBn);        // mat_W^T
  matmul(stream, Apack, Bpack, nullptr, nullptr, geae, MATn, nullptr, 0, En, EMBn, MATn, 0);
  pack_a(stream, gzr, EMBn, En, EMBn, Apack, EMBn);
  pack_b(stream, rbfW, EMBn, 1, EMBn, RBFDn, Bpack, EMBn);       // rbf_W^T
  matmul(stream, Apack, Bpack, nullptr, nullptr, grbe, RBFDn, nullptr, 0, En, EMBn, RBFDn, 0);

  // geometry backward
  k_clear<<<(En+255)/256, 256, 0, stream>>>(gd, En);
  k_clear<<<(Nn*3+255)/256, 256, 0, stream>>>(gpos, Nn*3);
  k_triplet_bwd<<<(Tn+127)/128, 128, 0, stream>>>(ai, aj, ak, pos, srcE, dB, angB, gsbf, gd, gpos);
  k_edge_geom_bwd<<<(En+63)/64, 64, 0, stream>>>(eidx, pos, ea, dB, envB, geae, grbe, gd, geat, gpos);

  // edge_attr_grad contraction + force assembly
  k_sgrads<<<Gn*An*3, 256, 0, stream>>>(eag, geat, sg);
  k_force<<<(Nn*3+255)/256, 256, 0, stream>>>(gpos, sg, out);
}
